// MVAE_IRLSTM_75041668596253
// MI455X (gfx1250) — matrix operand order repaired and checked by reading, not compiled, not run on hardware
//
#include <hip/hip_runtime.h>
#include <hip/hip_bf16.h>

// ---------------- model constants ----------------
#define Dm   1024
#define Hm   2048
#define OUTm 1021      // D - NC
#define Bm   256
#define Lm   64
#define PIm  3.14159265358979323846f

typedef __attribute__((ext_vector_type(16))) __bf16 v16bf;
typedef __attribute__((ext_vector_type(8)))  __bf16 v8bf;
typedef __attribute__((ext_vector_type(8)))  float  v8f;

static __device__ __forceinline__ float sigf(float x) { return 1.0f / (1.0f + expf(-x)); }

#define WMMA_BF16(A_, B_, C_) \
  __builtin_amdgcn_wmma_f32_16x16x32_bf16(false, (A_), false, (B_), (short)0, (C_), false, false)

#define SHUF16(lo, hi) \
  __builtin_shufflevector((lo), (hi), 0,1,2,3,4,5,6,7,8,9,10,11,12,13,14,15)

// ================= weight prep: transpose f32 [K,N] -> bf16 [Npad,K] ================
__global__ __launch_bounds__(256) void irlstm_transpose_w(
    const float* __restrict__ W, __bf16* __restrict__ WT,
    int K, int N, int Npad, int zero_diag)
{
  long i = (long)blockIdx.x * blockDim.x + threadIdx.x;
  long total = (long)Npad * K;
  if (i >= total) return;
  int n = (int)(i / K);
  int k = (int)(i % K);
  float v = (n < N) ? W[(long)k * N + n] : 0.0f;
  if (zero_diag && n == k) v = 0.0f;
  WT[i] = (__bf16)v;
}

__global__ __launch_bounds__(256) void irlstm_zero_f32(float* __restrict__ p, long n)
{
  long i = (long)blockIdx.x * blockDim.x + threadIdx.x;
  if (i < n) p[i] = 0.0f;
}

// ================= WMMA GEMM: C[M,N] = A[M,K](bf16) * WT[N,K](bf16)^T + bias =========
// wave32: each wave computes a 32x32 C tile as 2x2 WMMA 16x16 tiles.
__global__ __launch_bounds__(256) void irlstm_gemm_bf16(
    const __bf16* __restrict__ A, const __bf16* __restrict__ WT,
    const float* __restrict__ bias, int bias_len,
    float* __restrict__ C, int M, int N, int K)
{
  const int wave = threadIdx.x >> 5;
  const int lane = threadIdx.x & 31;
  const int ntn  = N >> 5;
  const int tile = blockIdx.x * 8 + wave;
  const int tm   = (tile / ntn) << 5;
  const int tn   = (tile % ntn) << 5;
  if (tm >= M) return;
  const int hs = lane >> 4;      // half-wave selector
  const int r  = lane & 15;      // row (A) / col (B) within 16

  const __bf16* a0p = A  + (size_t)(tm + r)      * K;
  const __bf16* a1p = A  + (size_t)(tm + 16 + r) * K;
  const __bf16* b0p = WT + (size_t)(tn + r)      * K;
  const __bf16* b1p = WT + (size_t)(tn + 16 + r) * K;

  v8f acc00 = {}, acc01 = {}, acc10 = {}, acc11 = {};

  for (int kk = 0; kk < K; kk += 32) {
    __builtin_prefetch(b0p + kk + 64, 0, 1);
    __builtin_prefetch(b1p + kk + 64, 0, 1);
    v8bf a0lo = *(const v8bf*)(a0p + kk + hs * 8);
    v8bf a0hi = *(const v8bf*)(a0p + kk + 16 + hs * 8);
    v8bf a1lo = *(const v8bf*)(a1p + kk + hs * 8);
    v8bf a1hi = *(const v8bf*)(a1p + kk + 16 + hs * 8);
    v16bf a0 = SHUF16(a0lo, a0hi);
    v16bf a1 = SHUF16(a1lo, a1hi);
    v16bf b0 = SHUF16(*(const v8bf*)(b0p + kk + hs * 8), *(const v8bf*)(b0p + kk + 16 + hs * 8));
    v16bf b1 = SHUF16(*(const v8bf*)(b1p + kk + hs * 8), *(const v8bf*)(b1p + kk + 16 + hs * 8));
    acc00 = WMMA_BF16(a0, b0, acc00);
    acc01 = WMMA_BF16(a0, b1, acc01);
    acc10 = WMMA_BF16(a1, b0, acc10);
    acc11 = WMMA_BF16(a1, b1, acc11);
  }

  const int c0 = tn + r, c1 = tn + 16 + r;
  const float bv0 = (bias && c0 < bias_len) ? bias[c0] : 0.0f;
  const float bv1 = (bias && c1 < bias_len) ? bias[c1] : 0.0f;
#pragma unroll
  for (int rr = 0; rr < 8; ++rr) {
    int row0 = tm + rr + 8 * hs;
    int row1 = tm + 16 + rr + 8 * hs;
    C[(size_t)row0 * N + c0] = acc00[rr] + bv0;
    C[(size_t)row0 * N + c1] = acc01[rr] + bv1;
    C[(size_t)row1 * N + c0] = acc10[rr] + bv0;
    C[(size_t)row1 * N + c1] = acc11[rr] + bv1;
  }
}

// ====== fused dual GEMM for gates: C = A1*W1T^T + A2*W2T^T + bias1 + bias2 ==========
__global__ __launch_bounds__(256) void irlstm_gemm2_bf16(
    const __bf16* __restrict__ A1, const __bf16* __restrict__ W1T, int K1,
    const __bf16* __restrict__ A2, const __bf16* __restrict__ W2T, int K2,
    const float* __restrict__ bias1, const float* __restrict__ bias2,
    float* __restrict__ C, int M, int N)
{
  const int wave = threadIdx.x >> 5;
  const int lane = threadIdx.x & 31;
  const int ntn  = N >> 5;
  const int tile = blockIdx.x * 8 + wave;
  const int tm   = (tile / ntn) << 5;
  const int tn   = (tile % ntn) << 5;
  if (tm >= M) return;
  const int hs = lane >> 4;
  const int r  = lane & 15;

  v8f acc00 = {}, acc01 = {}, acc10 = {}, acc11 = {};

  {
    const __bf16* a0p = A1  + (size_t)(tm + r)      * K1;
    const __bf16* a1p = A1  + (size_t)(tm + 16 + r) * K1;
    const __bf16* b0p = W1T + (size_t)(tn + r)      * K1;
    const __bf16* b1p = W1T + (size_t)(tn + 16 + r) * K1;
    for (int kk = 0; kk < K1; kk += 32) {
      __builtin_prefetch(b0p + kk + 64, 0, 1);
      __builtin_prefetch(b1p + kk + 64, 0, 1);
      v16bf a0 = SHUF16(*(const v8bf*)(a0p + kk + hs * 8), *(const v8bf*)(a0p + kk + 16 + hs * 8));
      v16bf a1 = SHUF16(*(const v8bf*)(a1p + kk + hs * 8), *(const v8bf*)(a1p + kk + 16 + hs * 8));
      v16bf b0 = SHUF16(*(const v8bf*)(b0p + kk + hs * 8), *(const v8bf*)(b0p + kk + 16 + hs * 8));
      v16bf b1 = SHUF16(*(const v8bf*)(b1p + kk + hs * 8), *(const v8bf*)(b1p + kk + 16 + hs * 8));
      acc00 = WMMA_BF16(a0, b0, acc00);
      acc01 = WMMA_BF16(a0, b1, acc01);
      acc10 = WMMA_BF16(a1, b0, acc10);
      acc11 = WMMA_BF16(a1, b1, acc11);
    }
  }
  {
    const __bf16* a0p = A2  + (size_t)(tm + r)      * K2;
    const __bf16* a1p = A2  + (size_t)(tm + 16 + r) * K2;
    const __bf16* b0p = W2T + (size_t)(tn + r)      * K2;
    const __bf16* b1p = W2T + (size_t)(tn + 16 + r) * K2;
    for (int kk = 0; kk < K2; kk += 32) {
      __builtin_prefetch(b0p + kk + 64, 0, 1);
      __builtin_prefetch(b1p + kk + 64, 0, 1);
      v16bf a0 = SHUF16(*(const v8bf*)(a0p + kk + hs * 8), *(const v8bf*)(a0p + kk + 16 + hs * 8));
      v16bf a1 = SHUF16(*(const v8bf*)(a1p + kk + hs * 8), *(const v8bf*)(a1p + kk + 16 + hs * 8));
      v16bf b0 = SHUF16(*(const v8bf*)(b0p + kk + hs * 8), *(const v8bf*)(b0p + kk + 16 + hs * 8));
      v16bf b1 = SHUF16(*(const v8bf*)(b1p + kk + hs * 8), *(const v8bf*)(b1p + kk + 16 + hs * 8));
      acc00 = WMMA_BF16(a0, b0, acc00);
      acc01 = WMMA_BF16(a0, b1, acc01);
      acc10 = WMMA_BF16(a1, b0, acc10);
      acc11 = WMMA_BF16(a1, b1, acc11);
    }
  }

  const int c0 = tn + r, c1 = tn + 16 + r;
  const float bv0 = bias1[c0] + bias2[c0];
  const float bv1 = bias1[c1] + bias2[c1];
#pragma unroll
  for (int rr = 0; rr < 8; ++rr) {
    int row0 = tm + rr + 8 * hs;
    int row1 = tm + 16 + rr + 8 * hs;
    C[(size_t)row0 * N + c0] = acc00[rr] + bv0;
    C[(size_t)row0 * N + c1] = acc01[rr] + bv1;
    C[(size_t)row1 * N + c0] = acc10[rr] + bv0;
    C[(size_t)row1 * N + c1] = acc11[rr] + bv1;
  }
}

// ================= elementwise kernels ==============================================

// t = 0: u = x[:,0], d = delta[:,0] (as bf16 fragments inputs)
__global__ __launch_bounds__(256) void irlstm_prep0(
    const float* __restrict__ x, const float* __restrict__ delta,
    __bf16* __restrict__ u_bf, __bf16* __restrict__ d_bf)
{
  int i = blockIdx.x * blockDim.x + threadIdx.x;        // over B*D
  if (i >= Bm * Dm) return;
  int b = i / Dm, d = i % Dm;
  long idx = ((long)b * Lm + 0) * Dm + d;
  u_bf[i] = (__bf16)x[idx];
  d_bf[i] = (__bf16)delta[idx];
}

// t >= 1 stage 1: gamma_x, x_c, gin = [gamma_x || m], d_bf
__global__ __launch_bounds__(256) void irlstm_impute1(
    const float* __restrict__ x, const float* __restrict__ m, const float* __restrict__ delta,
    const float* __restrict__ dgx_diag, const float* __restrict__ dgx_b,
    const float* __restrict__ x_hat,
    __bf16* __restrict__ xc_bf, __bf16* __restrict__ gin_bf, __bf16* __restrict__ d_bf, int t)
{
  int i = blockIdx.x * blockDim.x + threadIdx.x;        // over B*D
  if (i >= Bm * Dm) return;
  int b = i / Dm, d = i % Dm;
  long idx = ((long)b * Lm + t) * Dm + d;
  float xv = x[idx], mv = m[idx], dv = delta[idx];
  float gx = expf(-fmaxf(0.0f, dv * dgx_diag[d] + dgx_b[d]));
  float xc = mv * xv + (1.0f - mv) * x_hat[i];
  xc_bf[i] = (__bf16)xc;
  gin_bf[(long)b * (2 * Dm) + d]      = (__bf16)gx;
  gin_bf[(long)b * (2 * Dm) + Dm + d] = (__bf16)mv;
  d_bf[i] = (__bf16)dv;
}

// t >= 1 stage 2: beta gate blend -> u_t
__global__ __launch_bounds__(256) void irlstm_impute2(
    const float* __restrict__ x, const float* __restrict__ m,
    const float* __restrict__ zhat, const float* __restrict__ betapre,
    const float* __restrict__ x_hat, __bf16* __restrict__ u_bf, int t)
{
  int i = blockIdx.x * blockDim.x + threadIdx.x;        // over B*D
  if (i >= Bm * Dm) return;
  int b = i / Dm, d = i % Dm;
  long idx = ((long)b * Lm + t) * Dm + d;
  float beta = sigf(betapre[i]);
  float uh = beta * zhat[i] + (1.0f - beta) * x_hat[i];
  float u = m[idx] * x[idx] + (1.0f - m[idx]) * uh;
  u_bf[i] = (__bf16)u;
}

// gamma_h = exp(-relu(ghpre)); hhat = gamma_h * h  (ghpre already has dgh_b added)
__global__ __launch_bounds__(256) void irlstm_gammah(
    const float* __restrict__ ghpre, const float* __restrict__ h,
    __bf16* __restrict__ hhat_bf)
{
  int i = blockIdx.x * blockDim.x + threadIdx.x;        // over B*H
  if (i >= Bm * Hm) return;
  float g = expf(-fmaxf(0.0f, ghpre[i]));
  hhat_bf[i] = (__bf16)(g * h[i]);
}

// LSTM-like cell update
__global__ __launch_bounds__(256) void irlstm_cell(
    const float* __restrict__ gates, float* __restrict__ c, float* __restrict__ h,
    __bf16* __restrict__ h_bf)
{
  int i = blockIdx.x * blockDim.x + threadIdx.x;        // over B*H
  if (i >= Bm * Hm) return;
  int b = i / Hm, k = i % Hm;
  const float* gr_ = gates + (long)b * (4 * Hm);
  float f  = sigf(gr_[k]);
  float r  = sigf(gr_[Hm + k]);
  float cg = gr_[2 * Hm + k];
  float o  = gr_[3 * Hm + k];
  float g  = f + sinf(f * PIm) * cosf(r * PIm) / PIm;
  float chat = tanhf(cg);
  float cn = g * c[i] + (1.0f - g) * chat;
  float hn = sigf(o) * tanhf(cn);
  c[i] = cn;
  h[i] = hn;
  h_bf[i] = (__bf16)hn;
}

// write preds[b,t,:] = [xp || dxp] ; x_hat = [xp || softmax(dxp)]
__global__ __launch_bounds__(256) void irlstm_finalize(
    const float* __restrict__ xp, const float* __restrict__ dxp,
    float* __restrict__ x_hat, float* __restrict__ out, int t)
{
  int i = blockIdx.x * blockDim.x + threadIdx.x;        // over B*D
  if (i >= Bm * Dm) return;
  int b = i / Dm, j = i % Dm;
  float val;
  if (j < OUTm) {
    val = xp[(long)b * 1024 + j];
    x_hat[i] = val;
  } else {
    const float* dp = dxp + (long)b * 32;
    float v0 = dp[0], v1 = dp[1], v2 = dp[2];
    val = dp[j - OUTm];
    float mx = fmaxf(v0, fmaxf(v1, v2));
    float e0 = expf(v0 - mx), e1 = expf(v1 - mx), e2 = expf(v2 - mx);
    x_hat[i] = expf(val - mx) / (e0 + e1 + e2);
  }
  out[((long)b * Lm + t) * Dm + j] = val;
}

// ================= host orchestration ===============================================
extern "C" void kernel_launch(void* const* d_in, const int* in_sizes, int n_in,
                              void* d_out, int out_size, void* d_ws, size_t ws_size,
                              hipStream_t stream)
{
  const float* x        = (const float*)d_in[0];
  const float* m        = (const float*)d_in[1];
  const float* delta    = (const float*)d_in[2];
  const float* dgx_diag = (const float*)d_in[3];
  const float* dgx_b    = (const float*)d_in[4];
  const float* xz_w     = (const float*)d_in[5];
  const float* xz_b     = (const float*)d_in[6];
  const float* bdg_w    = (const float*)d_in[7];
  const float* bdg_b    = (const float*)d_in[8];
  const float* dgh_w    = (const float*)d_in[9];
  const float* dgh_b    = (const float*)d_in[10];
  const float* Wu       = (const float*)d_in[11];
  const float* bu       = (const float*)d_in[12];
  const float* Wh       = (const float*)d_in[13];
  const float* bh       = (const float*)d_in[14];
  const float* gh_w     = (const float*)d_in[15];
  const float* gh_b     = (const float*)d_in[16];
  const float* dxh_w    = (const float*)d_in[17];
  const float* dxh_b    = (const float*)d_in[18];
  float* out = (float*)d_out;

  // ---- workspace bump allocator ----
  char* wp = (char*)d_ws;
  auto alloc = [&](size_t bytes) -> void* {
    void* p = (void*)wp;
    wp += (bytes + 255) & ~(size_t)255;
    return p;
  };
  // bf16 transposed weights [N,K]
  __bf16* xzT  = (__bf16*)alloc((size_t)1024 * 1024 * 2);
  __bf16* bdgT = (__bf16*)alloc((size_t)1024 * 2048 * 2);
  __bf16* dghT = (__bf16*)alloc((size_t)2048 * 1024 * 2);
  __bf16* WuT  = (__bf16*)alloc((size_t)8192 * 1024 * 2);
  __bf16* WhT  = (__bf16*)alloc((size_t)8192 * 2048 * 2);
  __bf16* ghT  = (__bf16*)alloc((size_t)1024 * 2048 * 2);
  __bf16* dxhT = (__bf16*)alloc((size_t)32   * 2048 * 2);
  // f32 state / intermediates
  float* hS      = (float*)alloc((size_t)Bm * Hm * 4);
  float* cS      = (float*)alloc((size_t)Bm * Hm * 4);
  float* x_hat   = (float*)alloc((size_t)Bm * Dm * 4);
  float* gates   = (float*)alloc((size_t)Bm * 4 * Hm * 4);
  float* ghpre   = (float*)alloc((size_t)Bm * Hm * 4);
  float* zhat    = (float*)alloc((size_t)Bm * Dm * 4);
  float* betapre = (float*)alloc((size_t)Bm * Dm * 4);
  float* xp      = (float*)alloc((size_t)Bm * 1024 * 4);
  float* dxp     = (float*)alloc((size_t)Bm * 32 * 4);
  // bf16 activations
  __bf16* u_bf    = (__bf16*)alloc((size_t)Bm * Dm * 2);
  __bf16* d_bf    = (__bf16*)alloc((size_t)Bm * Dm * 2);
  __bf16* xc_bf   = (__bf16*)alloc((size_t)Bm * Dm * 2);
  __bf16* gin_bf  = (__bf16*)alloc((size_t)Bm * 2 * Dm * 2);
  __bf16* hhat_bf = (__bf16*)alloc((size_t)Bm * Hm * 2);
  __bf16* h_bf    = (__bf16*)alloc((size_t)Bm * Hm * 2);

  auto gblk = [](long n) { return dim3((unsigned)((n + 255) / 256)); };

  // ---- one-time (per call) weight prep: transpose + bf16 ----
  irlstm_transpose_w<<<gblk((long)1024 * 1024), 256, 0, stream>>>(xz_w,  xzT,  1024, 1024, 1024, 1);
  irlstm_transpose_w<<<gblk((long)1024 * 2048), 256, 0, stream>>>(bdg_w, bdgT, 2048, 1024, 1024, 0);
  irlstm_transpose_w<<<gblk((long)2048 * 1024), 256, 0, stream>>>(dgh_w, dghT, 1024, 2048, 2048, 0);
  irlstm_transpose_w<<<gblk((long)8192 * 1024), 256, 0, stream>>>(Wu,    WuT,  1024, 8192, 8192, 0);
  irlstm_transpose_w<<<gblk((long)8192 * 2048), 256, 0, stream>>>(Wh,    WhT,  2048, 8192, 8192, 0);
  irlstm_transpose_w<<<gblk((long)1024 * 2048), 256, 0, stream>>>(gh_w,  ghT,  2048, 1021, 1024, 0);
  irlstm_transpose_w<<<gblk((long)32 * 2048),   256, 0, stream>>>(dxh_w, dxhT, 2048, 3,    32,   0);

  irlstm_zero_f32<<<gblk((long)Bm * Hm), 256, 0, stream>>>(hS, (long)Bm * Hm);
  irlstm_zero_f32<<<gblk((long)Bm * Hm), 256, 0, stream>>>(cS, (long)Bm * Hm);

  const dim3 ew_d(gblk((long)Bm * Dm));   // elementwise over B*D
  const dim3 ew_h(gblk((long)Bm * Hm));   // elementwise over B*H

  // shared tail of every timestep (after u_bf / d_bf are ready)
  auto step_tail = [&](int t) {
    // ghpre = d @ dgh_w + dgh_b   [B,H]
    irlstm_gemm_bf16<<<Hm / 32, 256, 0, stream>>>(d_bf, dghT, dgh_b, Hm, ghpre, Bm, Hm, Dm);
    irlstm_gammah<<<ew_h, 256, 0, stream>>>(ghpre, hS, hhat_bf);
    // gates = u@Wu + hhat@Wh + bu + bh   [B,4H]
    irlstm_gemm2_bf16<<<(4 * Hm) / 32, 256, 0, stream>>>(u_bf, WuT, Dm, hhat_bf, WhT, Hm,
                                                         bu, bh, gates, Bm, 4 * Hm);
    irlstm_cell<<<ew_h, 256, 0, stream>>>(gates, cS, hS, h_bf);
    // xp = h @ gh_w + gh_b (N padded to 1024), dxp = h @ dxh_w + dxh_b (N padded to 32)
    irlstm_gemm_bf16<<<1024 / 32, 256, 0, stream>>>(h_bf, ghT, gh_b, OUTm, xp, Bm, 1024, Hm);
    irlstm_gemm_bf16<<<32 / 32, 256, 0, stream>>>(h_bf, dxhT, dxh_b, 3, dxp, Bm, 32, Hm);
    irlstm_finalize<<<ew_d, 256, 0, stream>>>(xp, dxp, x_hat, out, t);
  };

  // ---- t = 0: no imputation ----
  irlstm_prep0<<<ew_d, 256, 0, stream>>>(x, delta, u_bf, d_bf);
  step_tail(0);

  // ---- t = 1..L-1: impute then cell ----
  for (int t = 1; t < Lm; ++t) {
    irlstm_impute1<<<ew_d, 256, 0, stream>>>(x, m, delta, dgx_diag, dgx_b, x_hat,
                                             xc_bf, gin_bf, d_bf, t);
    // zhat = x_c @ xz_w(diag-zeroed) + xz_b     [B,D], K=D
    irlstm_gemm_bf16<<<Dm / 32, 256, 0, stream>>>(xc_bf, xzT, xz_b, Dm, zhat, Bm, Dm, Dm);
    // betapre = [gamma_x||m] @ bdg_w + bdg_b    [B,D], K=2D
    irlstm_gemm_bf16<<<Dm / 32, 256, 0, stream>>>(gin_bf, bdgT, bdg_b, Dm, betapre, Bm, Dm, 2 * Dm);
    irlstm_impute2<<<ew_d, 256, 0, stream>>>(x, m, zhat, betapre, x_hat, u_bf, t);
    step_tail(t);
  }
}
